// GNN_70978629534135
// MI455X (gfx1250) — matrix operand order repaired and checked by reading, compiled, not run on hardware
//
#include <hip/hip_runtime.h>
#include <hip/hip_bf16.h>
#include <math.h>

// ---------------------------------------------------------------------------
// GNN forward on MI455X (gfx1250): WMMA bf16 GEMMs + atomic scatter GCN hops.
// Block tile 128x64, 8 waves (4x2), 4 accumulators/wave -> 4 WMMAs per K-step.
// ---------------------------------------------------------------------------

typedef __attribute__((ext_vector_type(16))) __bf16 v16bf;
typedef __attribute__((ext_vector_type(8)))  __bf16 v8bf;
typedef __attribute__((ext_vector_type(8)))  float  v8f;

#define N_NODES 100000
#define N_EDGES 600000
#define F_IN    128
#define H_ENC   256
#define DIM     128
#define N_CLS   40
#define R_HOPS  3

__device__ __forceinline__ unsigned short f32_to_bf16_rne(float f) {
  union { float f; unsigned int u; } v; v.f = f;
  unsigned int u = v.u;
  unsigned int r = u + 0x7fffu + ((u >> 16) & 1u);   // round-to-nearest-even
  return (unsigned short)(r >> 16);
}

// ---- weight conversion + transpose: W[K][M] f32 -> Wt[M][K] bf16 -----------
__global__ void cvt_transpose_bf16(const float* __restrict__ W,
                                   unsigned short* __restrict__ Wt,
                                   int K, int M) {
  int i = blockIdx.x * blockDim.x + threadIdx.x;
  if (i < K * M) {
    int m = i / K;
    int k = i - m * K;
    Wt[i] = f32_to_bf16_rne(W[(size_t)k * M + m]);
  }
}

// ---- WMMA GEMM: C = act(A[f32, Nr x K] @ W + bias), W given as Wt[M][K] ----
template <bool RELU, bool HAS_BIAS>
__global__ __launch_bounds__(256)
void gemm_bf16_wmma(const float* __restrict__ A,
                    const unsigned short* __restrict__ Wt,  // [M][K] bf16
                    const float* __restrict__ bias,
                    float* __restrict__ C,
                    int Nr, int K, int M) {
  __shared__ __align__(16) unsigned short As[128 * 40];  // [row][k], stride 40

  const int t    = threadIdx.x;
  const int lane = t & 31;
  const int wid  = t >> 5;
  const int wm   = wid & 3;        // wave row group 0..3 (32 rows each)
  const int wn   = wid >> 2;       // wave col group 0..1 (32 cols each)
  const int rowBase = blockIdx.x * 128;
  const int colBase = blockIdx.y * 64;
  const int half = lane >> 4;
  const int l16  = lane & 15;

  // ---- loop-invariant addressing ----
  // A staging: this thread's 4 (row, c4) chunks, rows clamped (branch-free;
  // OOB rows only feed OOB outputs).
  const float* aptr[4];
  unsigned int  soff[4];
#pragma unroll
  for (int i = 0; i < 4; ++i) {
    int idx = i * 256 + t;              // 1024 float4 chunks per K-step
    int r   = idx >> 3;
    int c4  = (idx & 7) * 4;
    int gr  = rowBase + r;
    if (gr > Nr - 1) gr = Nr - 1;
    aptr[i] = A + (size_t)gr * K + c4;
    soff[i] = r * 40 + c4;
  }
  // B fragments: Wt row = output column -> two 8-K runs per lane-half (ISA table order).
  const unsigned short* bptr[2];
#pragma unroll
  for (int cg = 0; cg < 2; ++cg) {
    int gc = colBase + wn * 32 + cg * 16 + l16;
    if (gc > M - 1) gc = M - 1;         // garbage only in unstored columns
    bptr[cg] = Wt + (size_t)gc * K + half * 8;
  }
  // A fragments (ISA 16-bit 16x32 layout): two contiguous 16B runs per lane.
  const unsigned short* afrag[2][2];
#pragma unroll
  for (int rg = 0; rg < 2; ++rg) {
    const unsigned short* ar = &As[(wm * 32 + rg * 16 + l16) * 40];
    afrag[rg][0] = ar + half * 8;
    afrag[rg][1] = ar + 16 + half * 8;
  }

  v8f acc[2][2] = {};

  for (int k0 = 0; k0 < K; k0 += 32) {
    // stage A tile (128x32) f32 -> bf16: b128 global loads, b64 LDS stores
#pragma unroll
    for (int i = 0; i < 4; ++i) {
      const float4 v = *(const float4*)(aptr[i] + k0);
      unsigned int p0 = (unsigned int)f32_to_bf16_rne(v.x) |
                        ((unsigned int)f32_to_bf16_rne(v.y) << 16);
      unsigned int p1 = (unsigned int)f32_to_bf16_rne(v.z) |
                        ((unsigned int)f32_to_bf16_rne(v.w) << 16);
      uint2 q; q.x = p0; q.y = p1;
      *(uint2*)(&As[soff[i]]) = q;
    }

    // B fragments straight from global (two b128 loads each)
    v16bf bfr[2];
#pragma unroll
    for (int cg = 0; cg < 2; ++cg) {
      const v8bf* pb = (const v8bf*)(bptr[cg] + k0);
      bfr[cg] = __builtin_shufflevector(pb[0], pb[2],
          0, 1, 2, 3, 4, 5, 6, 7, 8, 9, 10, 11, 12, 13, 14, 15);
    }

    __syncthreads();

#pragma unroll
    for (int rg = 0; rg < 2; ++rg) {
      const v8bf* pa0 = (const v8bf*)afrag[rg][0];
      const v8bf* pa1 = (const v8bf*)afrag[rg][1];
      v16bf a = __builtin_shufflevector(pa0[0], pa1[0],
          0, 1, 2, 3, 4, 5, 6, 7, 8, 9, 10, 11, 12, 13, 14, 15);
#pragma unroll
      for (int cg = 0; cg < 2; ++cg) {
        acc[rg][cg] = __builtin_amdgcn_wmma_f32_16x16x32_bf16(
            false, a, false, bfr[cg], (short)0, acc[rg][cg], false, false);
      }
    }
    __syncthreads();
  }

  // ---- epilogue: C layout — VGPR v: row = v + 8*half, col = lane%16.
  // Fast path for full row blocks (all but the grid tail).
  const bool fullRows = (rowBase + 128 <= Nr);
#pragma unroll
  for (int cg = 0; cg < 2; ++cg) {
    const int col = colBase + wn * 32 + cg * 16 + l16;
    if (col >= M) continue;
    const float bv = HAS_BIAS ? bias[col] : 0.0f;
#pragma unroll
    for (int rg = 0; rg < 2; ++rg) {
      const int row0 = rowBase + wm * 32 + rg * 16 + half * 8;
      float* cp = C + (size_t)row0 * M + col;
      if (fullRows) {
#pragma unroll
        for (int v = 0; v < 8; ++v) {
          float xv = acc[rg][cg][v] + bv;
          if (RELU) xv = fmaxf(xv, 0.0f);
          *cp = xv;
          cp += M;
        }
      } else {
#pragma unroll
        for (int v = 0; v < 8; ++v) {
          if (row0 + v < Nr) {
            float xv = acc[rg][cg][v] + bv;
            if (RELU) xv = fmaxf(xv, 0.0f);
            *cp = xv;
          }
          cp += M;
        }
      }
    }
  }
}

// ---- GCN aggregation helpers ----------------------------------------------
__global__ void zero_f32(float* __restrict__ p, int n) {
  int i = blockIdx.x * blockDim.x + threadIdx.x;
  if (i < n) p[i] = 0.0f;
}

// One wave per edge: 32 lanes x float4 = one 128-float row per wave.
__global__ __launch_bounds__(256)
void scatter_add_edges(const float* __restrict__ hw,
                       const int* __restrict__ src,
                       const int* __restrict__ dst,
                       float* __restrict__ agg, int E) {
  long long idx = (long long)blockIdx.x * blockDim.x + threadIdx.x;
  if (idx >= (long long)E * 32) return;
  int e = (int)(idx >> 5);
  int f = (int)(idx & 31) * 4;
  int s = src[e];              // wave-uniform (scalar) loads
  int d = dst[e];
  const float4 v = *(const float4*)(hw + (long long)s * DIM + f);
  float* o = agg + (long long)d * DIM + f;
  atomicAdd(o + 0, v.x);
  atomicAdd(o + 1, v.y);
  atomicAdd(o + 2, v.z);
  atomicAdd(o + 3, v.w);
}

__global__ void bias_relu_inplace(float* __restrict__ h,
                                  const float* __restrict__ b, int n) {
  int i = blockIdx.x * blockDim.x + threadIdx.x;
  if (i < n) h[i] = fmaxf(h[i] + b[i & (DIM - 1)], 0.0f);
}

// ---- log-softmax over 40 classes: one wave per node -----------------------
__global__ __launch_bounds__(256)
void log_softmax_rows(float* __restrict__ out, int n) {
  int lane = threadIdx.x & 31;
  int wid  = threadIdx.x >> 5;
  int row  = blockIdx.x * 8 + wid;
  if (row >= n) return;
  float* p = out + (long long)row * N_CLS;
  float v1 = p[lane];
  float v2 = (lane < N_CLS - 32) ? p[32 + lane] : -INFINITY;
  float m = fmaxf(v1, v2);
#pragma unroll
  for (int o = 16; o > 0; o >>= 1) m = fmaxf(m, __shfl_xor(m, o, 32));
  float s = __expf(v1 - m) + ((lane < N_CLS - 32) ? __expf(v2 - m) : 0.0f);
#pragma unroll
  for (int o = 16; o > 0; o >>= 1) s += __shfl_xor(s, o, 32);
  float ls = __logf(s);
  p[lane] = v1 - m - ls;
  if (lane < N_CLS - 32) p[32 + lane] = v2 - m - ls;
}

// ---------------------------------------------------------------------------
extern "C" void kernel_launch(void* const* d_in, const int* in_sizes, int n_in,
                              void* d_out, int out_size, void* d_ws, size_t ws_size,
                              hipStream_t stream) {
  const float* x      = (const float*)d_in[0];
  const int*   eidx   = (const int*)  d_in[1];
  const float* enc_w1 = (const float*)d_in[2];
  const float* enc_b1 = (const float*)d_in[3];
  const float* enc_w2 = (const float*)d_in[4];
  const float* enc_b2 = (const float*)d_in[5];
  const float* gcn_w  = (const float*)d_in[6];
  const float* gcn_b  = (const float*)d_in[7];
  const float* dec_w1 = (const float*)d_in[8];
  const float* dec_b1 = (const float*)d_in[9];
  const float* dec_w2 = (const float*)d_in[10];
  const float* dec_b2 = (const float*)d_in[11];
  float* out = (float*)d_out;

  const int* src = eidx;            // edge_index[0]
  const int* dst = eidx + N_EDGES;  // edge_index[1]

  // workspace carve-up
  char* ws = (char*)d_ws;
  size_t off = 0;
  auto carve = [&](size_t bytes) -> char* {
    char* p = ws + off;
    off += (bytes + 255) & ~(size_t)255;
    return p;
  };
  unsigned short* w_enc1 = (unsigned short*)carve((size_t)F_IN * H_ENC * 2);
  unsigned short* w_enc2 = (unsigned short*)carve((size_t)H_ENC * DIM * 2);
  unsigned short* w_gcn  = (unsigned short*)carve((size_t)R_HOPS * DIM * DIM * 2);
  unsigned short* w_dec1 = (unsigned short*)carve((size_t)DIM * DIM * 2);
  unsigned short* w_dec2 = (unsigned short*)carve((size_t)DIM * N_CLS * 2);
  float* h256 = (float*)carve((size_t)N_NODES * H_ENC * sizeof(float));
  float* hA   = (float*)carve((size_t)N_NODES * DIM * sizeof(float));
  float* hB   = (float*)carve((size_t)N_NODES * DIM * sizeof(float));

  // convert + transpose weights to bf16 Wt[M][K]
  auto cvtT = [&](const float* in, unsigned short* o, int K, int M) {
    cvt_transpose_bf16<<<(K * M + 255) / 256, 256, 0, stream>>>(in, o, K, M);
  };
  cvtT(enc_w1, w_enc1, F_IN, H_ENC);
  cvtT(enc_w2, w_enc2, H_ENC, DIM);
  for (int r = 0; r < R_HOPS; ++r)
    cvtT(gcn_w + (size_t)r * DIM * DIM, w_gcn + (size_t)r * DIM * DIM, DIM, DIM);
  cvtT(dec_w1, w_dec1, DIM, DIM);
  cvtT(dec_w2, w_dec2, DIM, N_CLS);

  const int rowBlocks = (N_NODES + 127) / 128;

  // encoder: h256 = relu(x @ enc_w1 + b1); hA = h256 @ enc_w2 + b2
  gemm_bf16_wmma<true, true><<<dim3(rowBlocks, H_ENC / 64), 256, 0, stream>>>(
      x, w_enc1, enc_b1, h256, N_NODES, F_IN, H_ENC);
  gemm_bf16_wmma<false, true><<<dim3(rowBlocks, DIM / 64), 256, 0, stream>>>(
      h256, w_enc2, enc_b2, hA, N_NODES, H_ENC, DIM);

  // GCN hops: hw = hA @ W_r; agg(hA) = scatter_add(hw[src] -> dst); relu(+b)
  const int nElems = N_NODES * DIM;
  const long long scatterThreads = (long long)N_EDGES * 32;
  for (int r = 0; r < R_HOPS; ++r) {
    gemm_bf16_wmma<false, false><<<dim3(rowBlocks, DIM / 64), 256, 0, stream>>>(
        hA, w_gcn + (size_t)r * DIM * DIM, nullptr, hB, N_NODES, DIM, DIM);
    zero_f32<<<(nElems + 255) / 256, 256, 0, stream>>>(hA, nElems);
    scatter_add_edges<<<(int)((scatterThreads + 255) / 256), 256, 0, stream>>>(
        hB, src, dst, hA, N_EDGES);
    bias_relu_inplace<<<(nElems + 255) / 256, 256, 0, stream>>>(
        hA, gcn_b + (size_t)r * DIM, nElems);
  }

  // decoder: hB = relu(hA @ dec_w1 + b1); out = hB @ dec_w2 + b2
  gemm_bf16_wmma<true, true><<<dim3(rowBlocks, DIM / 64), 256, 0, stream>>>(
      hA, w_dec1, dec_b1, hB, N_NODES, DIM, DIM);
  gemm_bf16_wmma<false, true><<<dim3(rowBlocks, 1), 256, 0, stream>>>(
      hB, w_dec2, dec_b2, out, N_NODES, DIM, N_CLS);

  // log_softmax in place on d_out
  log_softmax_rows<<<(N_NODES + 7) / 8, 256, 0, stream>>>(out, N_NODES);
}
